// GraphAttention_76897094467735
// MI455X (gfx1250) — matrix operand order repaired and checked by reading, not compiled, not run on hardware
//
#include <hip/hip_runtime.h>

// GAT fused kernel set for gfx1250 (wave32, WMMA 16x16x32 f16).
// Roofline: adj (128 MB int32) dominates -> single pass over adj.
// Softmax uses an exact fixed per-row bound M_i = lrelu(s1_i + max_j s2_j)
// (softmax is shift-invariant), eliminating all online-rescale work.

typedef __attribute__((ext_vector_type(4)))  int      v4i;
typedef __attribute__((ext_vector_type(4)))  float    v4f;
typedef __attribute__((ext_vector_type(8)))  float    v8f;
typedef __attribute__((ext_vector_type(8)))  _Float16 v8h;
typedef __attribute__((ext_vector_type(16))) _Float16 v16h;

#define GAT_ALPHA (0.2f)

// ---- Kernel 0: W (f32, [128][64]) -> WT16 (f16, [64][128]) -----------------
__global__ void prep_wt_kernel(const float* __restrict__ W,
                               _Float16* __restrict__ wt) {
  int id = blockIdx.x * blockDim.x + threadIdx.x;
  if (id < 128 * 64) {
    int k = id >> 6;        // 0..127
    int f = id & 63;        // 0..63
    wt[f * 128 + k] = (_Float16)W[k * 64 + f];
  }
}

// ---- Kernel 1: Wh = x @ W via WMMA, stored transposed f16 ------------------
// 1 wave per 16-row tile of the flattened [B*N, 128] x matrix.
// Output: whT[b][f][n] (f16), f in [0,64), n in [0,2048).
__global__ void __launch_bounds__(128) gemm_wh_kernel(
    const float* __restrict__ x, const _Float16* __restrict__ wt,
    _Float16* __restrict__ whT) {
  const int lane = threadIdx.x & 31;
  const int wave = threadIdx.x >> 5;
  const int g    = blockIdx.x * 4 + wave;   // tile id, 1024 total
  const int row0 = g * 16;                  // flattened row base
  const int m    = lane & 15;               // A-row / C-col lane index
  const int kh   = lane >> 4;               // half-wave select

  const float* xr = x + (size_t)(row0 + m) * 128;

  v8f acc[4];
#pragma unroll
  for (int s = 0; s < 4; ++s) acc[s] = (v8f)0.0f;

#pragma unroll
  for (int kc = 0; kc < 128; kc += 32) {
    // A (16x32 f16): lane holds K runs [kc+8kh, +8) and [kc+16+8kh, +8)
    const v4f* p0 = (const v4f*)(xr + kc + 8 * kh);
    v4f x0 = p0[0], x1 = p0[1];
    const v4f* p1 = (const v4f*)(xr + kc + 16 + 8 * kh);
    v4f x2 = p1[0], x3 = p1[1];
    v16h a;
#pragma unroll
    for (int t = 0; t < 4; ++t) {
      a[t]      = (_Float16)x0[t];
      a[4 + t]  = (_Float16)x1[t];
      a[8 + t]  = (_Float16)x2[t];
      a[12 + t] = (_Float16)x3[t];
    }
#pragma unroll
    for (int s = 0; s < 4; ++s) {
      // B (32x16 f16): lane n=m holds K runs [kc+8kh, +8) and [kc+16+8kh, +8) of column 16s+m
      const v8h* bp = (const v8h*)(wt + (size_t)(16 * s + m) * 128 + kc + 8 * kh);
      v8h b0 = bp[0], b1 = bp[2];
      v16h bm;
#pragma unroll
      for (int t = 0; t < 8; ++t) { bm[t] = b0[t]; bm[8 + t] = b1[t]; }
      acc[s] = __builtin_amdgcn_wmma_f32_16x16x32_f16(
          false, a, false, bm, (short)0, acc[s], false, false);
    }
  }

  // C layout: acc[s][r] = Wh[row0 + r + 8*kh][16*s + m]; store transposed f16
  const int b   = row0 >> 11;     // / 2048 (tiles never straddle batches)
  const int nl0 = row0 & 2047;
#pragma unroll
  for (int s = 0; s < 4; ++s) {
#pragma unroll
    for (int r = 0; r < 8; ++r) {
      whT[((size_t)b * 64 + 16 * s + m) * 2048 + nl0 + r + 8 * kh] =
          (_Float16)acc[s][r];
    }
  }
}

// ---- Kernel 2: s1/s2 = Wh . a1 / Wh . a2 -----------------------------------
__global__ void s12_kernel(const _Float16* __restrict__ whT,
                           const float* __restrict__ a,
                           float* __restrict__ s1, float* __restrict__ s2) {
  int t = blockIdx.x * blockDim.x + threadIdx.x;   // 16384
  int b = t >> 11, n = t & 2047;
  const _Float16* base = whT + (size_t)b * 64 * 2048 + n;
  float acc1 = 0.f, acc2 = 0.f;
#pragma unroll 8
  for (int f = 0; f < 64; ++f) {
    float v = (float)base[(size_t)f * 2048];   // coalesced across lanes (n)
    acc1 += v * a[f];
    acc2 += v * a[64 + f];
  }
  s1[t] = acc1;
  s2[t] = acc2;
}

// ---- Kernel 2b: per-batch max of s2 (8 scalars) ----------------------------
__global__ void s2max_kernel(const float* __restrict__ s2,
                             float* __restrict__ s2max) {
  __shared__ float red[256];
  const int b = blockIdx.x;
  float mv = -3.0e38f;
  for (int t = threadIdx.x; t < 2048; t += 256)
    mv = fmaxf(mv, s2[b * 2048 + t]);
  red[threadIdx.x] = mv;
  __syncthreads();
  for (int s = 128; s > 0; s >>= 1) {
    if (threadIdx.x < s)
      red[threadIdx.x] = fmaxf(red[threadIdx.x], red[threadIdx.x + s]);
    __syncthreads();
  }
  if (threadIdx.x == 0) s2max[b] = red[0];
}

// ---- Kernel 3: fused masked-softmax attention + P@Wh -----------------------
// 1 wave per (batch, 16-row tile), producing 16x64 outputs. Single pass over
// adj; fixed per-row softmax bound -> no online rescale in the hot loop.
// 4x v_wmma_f32_16x16x32_f16 per 32-column chunk.
__global__ void __launch_bounds__(128) attn_kernel(
    const int* __restrict__ adj, const _Float16* __restrict__ whT,
    const float* __restrict__ s1g, const float* __restrict__ s2g,
    const float* __restrict__ s2maxg, float* __restrict__ out) {
  __shared__ float s2s[2048];

  const int b    = blockIdx.x >> 5;         // batch
  const int tg   = blockIdx.x & 31;         // tile group
  const int wave = threadIdx.x >> 5;
  const int lane = threadIdx.x & 31;
  const int m    = lane & 15;
  const int kh   = lane >> 4;
  const int i0   = (tg * 4 + wave) * 16;    // row-tile base within batch

  for (int t = threadIdx.x; t < 2048; t += 128) s2s[t] = s2g[b * 2048 + t];
  __syncthreads();

  const float s1v = s1g[b * 2048 + i0 + m];
  // Exact unmasked row max: lrelu is monotonic, so
  // max_j lrelu(s1+s2_j) = lrelu(s1 + max_j s2_j). Valid softmax shift.
  const float tb   = s1v + s2maxg[b];
  const float Mrow = fmaxf(tb, GAT_ALPHA * tb);

  const int*      adjr = adj + (size_t)(b * 2048 + i0 + m) * 2048;
  const _Float16* whb  = whT + (size_t)b * 64 * 2048;

  float rowl = 0.0f;      // lane-local partial row sum (combined at the end)
  v8f acc[4];
#pragma unroll
  for (int s = 0; s < 4; ++s) acc[s] = (v8f)0.0f;

  for (int jc = 0; jc < 2048; jc += 32) {
    if (jc + 64 < 2048)
      __builtin_prefetch(adjr + jc + 64, 0, 1);   // global_prefetch_b8

    // adj 16x32 tile, directly in WMMA-A element order per lane
    const v4i* ap0 = (const v4i*)(adjr + jc + 8 * kh);
    v4i ad0 = ap0[0], ad1 = ap0[1];
    const v4i* ap1 = (const v4i*)(adjr + jc + 16 + 8 * kh);
    v4i ad2 = ap1[0], ad3 = ap1[1];
    // matching s2 values from LDS
    const v4f* sp0 = (const v4f*)(s2s + jc + 8 * kh);
    v4f sA = sp0[0], sB = sp0[1];
    const v4f* sp1 = (const v4f*)(s2s + jc + 16 + 8 * kh);
    v4f sC = sp1[0], sD = sp1[1];

    v16h pa;
    float psum = 0.f;
#pragma unroll
    for (int t = 0; t < 4; ++t) {
      float t0 = s1v + sA[t]; t0 = fmaxf(t0, GAT_ALPHA * t0);
      float p0 = __expf(t0 - Mrow); p0 = (ad0[t] > 0) ? p0 : 0.0f;
      psum += p0; pa[t] = (_Float16)p0;

      float t1 = s1v + sB[t]; t1 = fmaxf(t1, GAT_ALPHA * t1);
      float p1 = __expf(t1 - Mrow); p1 = (ad1[t] > 0) ? p1 : 0.0f;
      psum += p1; pa[4 + t] = (_Float16)p1;

      float t2 = s1v + sC[t]; t2 = fmaxf(t2, GAT_ALPHA * t2);
      float p2 = __expf(t2 - Mrow); p2 = (ad2[t] > 0) ? p2 : 0.0f;
      psum += p2; pa[8 + t] = (_Float16)p2;

      float t3 = s1v + sD[t]; t3 = fmaxf(t3, GAT_ALPHA * t3);
      float p3 = __expf(t3 - Mrow); p3 = (ad3[t] > 0) ? p3 : 0.0f;
      psum += p3; pa[12 + t] = (_Float16)p3;
    }
    rowl += psum;

    // P (16x32) @ Wh-chunk (32x64) via 4 WMMAs, f32 accumulate
#pragma unroll
    for (int s = 0; s < 4; ++s) {
      const v8h* bp =
          (const v8h*)(whb + (size_t)(16 * s + m) * 2048 + jc + 8 * kh);
      v8h b0 = bp[0], b1 = bp[2];
      v16h bm;
#pragma unroll
      for (int t = 0; t < 8; ++t) { bm[t] = b0[t]; bm[8 + t] = b1[t]; }
      acc[s] = __builtin_amdgcn_wmma_f32_16x16x32_f16(
          false, pa, false, bm, (short)0, acc[s], false, false);
    }
  }

  // Combine the two half-wave partial sums once, then broadcast per row.
  rowl += __shfl_xor(rowl, 16);
  float linv = (rowl > 0.f) ? (1.0f / rowl) : 0.0f;
  float lb[8];
#pragma unroll
  for (int r = 0; r < 8; ++r) lb[r] = __shfl(linv, r + 8 * kh);
#pragma unroll
  for (int s = 0; s < 4; ++s)
#pragma unroll
    for (int r = 0; r < 8; ++r)
      out[(size_t)(b * 2048 + i0 + r + 8 * kh) * 64 + 16 * s + m] =
          acc[s][r] * lb[r];
}

// ---- Host-side launcher ----------------------------------------------------
extern "C" void kernel_launch(void* const* d_in, const int* in_sizes, int n_in,
                              void* d_out, int out_size, void* d_ws,
                              size_t ws_size, hipStream_t stream) {
  (void)in_sizes; (void)n_in; (void)out_size; (void)ws_size;
  const float* x   = (const float*)d_in[0];   // [8][2048][128] f32
  const int*   adj = (const int*)d_in[1];     // [8][2048][2048] i32
  const float* W   = (const float*)d_in[2];   // [128][64] f32
  const float* a   = (const float*)d_in[3];   // [128][1] f32
  float*       out = (float*)d_out;           // [8][2048][64] f32

  char* ws = (char*)d_ws;
  _Float16* whT  = (_Float16*)ws;                                // 2 MB
  _Float16* wt   = (_Float16*)(ws + (size_t)(2u << 20));         // 16 KB
  float*    s1   = (float*)(ws + (size_t)(2u << 20) + (1u << 16));
  float*    s2   = (float*)(ws + (size_t)(2u << 20) + (2u << 16));
  float*    s2mx = (float*)(ws + (size_t)(2u << 20) + (3u << 16));

  prep_wt_kernel<<<32, 256, 0, stream>>>(W, wt);
  gemm_wh_kernel<<<256, 128, 0, stream>>>(x, wt, whT);
  s12_kernel<<<64, 256, 0, stream>>>(whT, a, s1, s2);
  s2max_kernel<<<8, 256, 0, stream>>>(s2, s2mx);
  attn_kernel<<<256, 128, 0, stream>>>(adj, whT, s1, s2, s2mx, out);
}
